// DQMITBF_MultiScale_64029372449489
// MI455X (gfx1250) — matrix operand order repaired and checked by reading, not compiled, not run on hardware
//
#include <hip/hip_runtime.h>

// ---------------------------------------------------------------------------
// Types for WMMA fragments (avoid HIP vector classes inside unions)
// ---------------------------------------------------------------------------
typedef __attribute__((ext_vector_type(16))) __bf16  v16bf;
typedef __attribute__((ext_vector_type(8)))  float   v8f;
typedef __attribute__((ext_vector_type(4)))  unsigned int u32x4;
typedef __attribute__((ext_vector_type(4)))  int      v4i;

union Frag {
    u32x4 u[2];
    v16bf v;
};

// ---------------------------------------------------------------------------
// CDNA5 async global->LDS path (ASYNCcnt), with sync fallback
// ---------------------------------------------------------------------------
#if defined(__HIP_DEVICE_COMPILE__) && __has_builtin(__builtin_amdgcn_global_load_async_to_lds_b128)
#define HAVE_ASYNC_LDS 1
#else
#define HAVE_ASYNC_LDS 0
#endif

typedef __attribute__((address_space(1))) v4i* as1_v4i;   // global int4*
typedef __attribute__((address_space(3))) v4i* as3_v4i;   // LDS int4*

__device__ __forceinline__ void async_copy_b128(const __bf16* g, __bf16* l) {
#if HAVE_ASYNC_LDS
    // LDS generic pointer -> 32-bit offset == AS(3) value (aperture scheme:
    // LDS offset is the low 32 address bits); global generic == AS(1) numerically.
    __builtin_amdgcn_global_load_async_to_lds_b128(
        (as1_v4i)(unsigned long long)g,
        (as3_v4i)(unsigned)(unsigned long long)l,
        0, 0);
#else
    *(u32x4*)l = *(const u32x4*)g;
#endif
}

__device__ __forceinline__ void wait_async() {
#if HAVE_ASYNC_LDS
#if __has_builtin(__builtin_amdgcn_s_wait_asynccnt)
    __builtin_amdgcn_s_wait_asynccnt(0);
#else
    asm volatile("s_wait_asynccnt 0x0" ::: "memory");
#endif
#endif
}

__device__ __forceinline__ v8f wmma_bf16(v16bf a, v16bf b, v8f c) {
    // D = A(16x32 bf16) * B(32x16 bf16) + C(16x16 f32)
    return __builtin_amdgcn_wmma_f32_16x16x32_bf16(
        /*neg_a=*/false, a, /*neg_b=*/false, b,
        /*c_mod=*/(short)0, c, /*reuse_a=*/false, /*reuse_b=*/false);
}

__device__ __forceinline__ __bf16 to_bf16(float f) {
    unsigned u = __builtin_bit_cast(unsigned, f);
    u += 0x7fffu + ((u >> 16) & 1u);          // round-to-nearest-even
    unsigned short h = (unsigned short)(u >> 16);
    return __builtin_bit_cast(__bf16, h);
}

// ---------------------------------------------------------------------------
// Prep: transpose+cast  dst[N][C] (bf16) from channel-major fp32 (rows split
// between pointers a (c < halfC) and b (c >= halfC)).
// ---------------------------------------------------------------------------
__global__ void tcast_k(__bf16* __restrict__ dst,
                        const float* __restrict__ a, const float* __restrict__ b,
                        int halfC, int C, int N) {
    __shared__ float tile[32][33];
    const int bx = blockIdx.x * 32;   // along N
    const int by = blockIdx.y * 32;   // along C
    const int tx = threadIdx.x, ty = threadIdx.y;
#pragma unroll
    for (int i = 0; i < 4; ++i) {
        int c = by + ty + i * 8;
        const float* row = (c < halfC) ? (a + (size_t)c * N)
                                       : (b + (size_t)(c - halfC) * N);
        tile[ty + i * 8][tx] = row[bx + tx];
    }
    __syncthreads();
#pragma unroll
    for (int i = 0; i < 4; ++i) {
        int n = bx + ty + i * 8;
        dst[(size_t)n * C + by + tx] = to_bf16(tile[tx][ty + i * 8]);
    }
}

// Straight cast (same layout): Vc[c][m] = bf16(cat[c][m])
__global__ void ccast_k(__bf16* __restrict__ dst,
                        const float* __restrict__ a, const float* __restrict__ b,
                        int halfCN, int total) {
    for (int i = blockIdx.x * blockDim.x + threadIdx.x; i < total;
         i += gridDim.x * blockDim.x) {
        float v = (i < halfCN) ? a[i] : b[i - halfCN];
        dst[i] = to_bf16(v);
    }
}

// ---------------------------------------------------------------------------
// Flash attention: out[n][c] = softmax_m( Qt[n]·Kt[m] ) · V
//   Qt: [N][C] bf16, Kt: [N][C] bf16 (key-major), Vc: [C][N] bf16
//   Wave handles 16 query rows; wave-pair splits output channels in half.
//   Double-buffered async global->LDS staging of K/V tiles.
// ---------------------------------------------------------------------------
template <int C, int QG, int BN, bool VSMEM>
__global__ void __launch_bounds__(QG * 64)
flash_attn(const __bf16* __restrict__ Qt, const __bf16* __restrict__ Kt,
           const __bf16* __restrict__ Vc, float* __restrict__ outb, int N) {
    constexpr int KTP = C + 8;        // padded K-tile row pitch (bank spread)
    constexpr int VCP = BN + 8;       // padded V-tile / P row pitch
    constexpr int NJ = BN / 16;       // score column tiles per key tile
    constexpr int NT = (C / 2) / 16;  // output channel tiles per wave
    constexpr int NK = C / 32;        // k-chunks for QK^T
    constexpr int NTHR = QG * 64;
    constexpr int KTSZ = BN * KTP;    // one K-tile buffer (elems)
    constexpr int VTSZ = C * VCP;     // one V-tile buffer (elems)

    extern __shared__ __bf16 smem[];
    // [Kt buf0][Kt buf1][Vc buf0][Vc buf1][P]
    __bf16* P_s = smem + 2 * KTSZ + (VSMEM ? 2 * VTSZ : 0); // [QG][16][VCP]

    const int tid   = threadIdx.x;
    const int wave  = tid >> 5;
    const int lane  = tid & 31;
    const int qg    = wave >> 1;
    const int chalf = wave & 1;
    const int col   = lane & 15;
    const int hi    = lane >> 4;
    const int n0    = blockIdx.x * (QG * 16) + qg * 16;
    const int c0o   = chalf * (C / 2);

    v8f O[NT] = {};
    float m_st[8], l_st[8];
#pragma unroll
    for (int v = 0; v < 8; ++v) { m_st[v] = -1e30f; l_st[v] = 0.f; }

    const __bf16* qrow = Qt + (size_t)(n0 + col) * C;   // this lane's A row
    __bf16* pb = P_s + (size_t)qg * 16 * VCP;

    // issue async loads of one K/V tile into LDS buffer `buf`
    auto stage_tile = [&](int mb, int buf) {
        __bf16* ktb = smem + buf * KTSZ;
        const __bf16* gk = Kt + (size_t)mb * C;
        for (int i = tid; i < BN * (C / 8); i += NTHR) {
            int r = i / (C / 8), o = i % (C / 8);
            async_copy_b128(gk + (size_t)r * C + o * 8, ktb + r * KTP + o * 8);
        }
        if (VSMEM) {
            __bf16* vtb = smem + 2 * KTSZ + buf * VTSZ;
            for (int i = tid; i < C * (BN / 8); i += NTHR) {
                int c = i / (BN / 8), o = i % (BN / 8);
                async_copy_b128(Vc + (size_t)c * N + mb + o * 8,
                                vtb + c * VCP + o * 8);
            }
        }
    };

    stage_tile(0, 0);                         // prologue: tile 0 in flight

    const int T = N / BN;
    for (int it = 0; it < T; ++it) {
        const int mb  = it * BN;
        const int buf = it & 1;
        wait_async();                         // my async loads for tile `it`
        __syncthreads();                      // everyone's loads + prev compute
        if (it + 1 < T) stage_tile(mb + BN, buf ^ 1);  // overlap next tile

        const __bf16* Kt_b = smem + buf * KTSZ;
        const __bf16* Vc_b = smem + 2 * KTSZ + buf * VTSZ;

        // ---- S = Q_tile * K_tile^T   (16 x BN per wave)
        v8f S[NJ] = {};
#pragma unroll
        for (int kk = 0; kk < NK; ++kk) {
            Frag A;
            A.u[0] = *(const u32x4*)(qrow + kk * 32 + hi * 8);
            A.u[1] = *(const u32x4*)(qrow + kk * 32 + 16 + hi * 8);
#pragma unroll
            for (int j = 0; j < NJ; ++j) {
                Frag B;
                const __bf16* krow = Kt_b + (size_t)(j * 16 + col) * KTP
                                   + kk * 32;
                B.u[0] = *(const u32x4*)(krow + hi * 8);
                B.u[1] = *(const u32x4*)(krow + 16 + hi * 8);
                S[j] = wmma_bf16(A.v, B.v, S[j]);
            }
        }

        // ---- online softmax (rows live as: VGPR v -> row v + 8*hi)
        float P2[NJ][8];
        float scl[8];
#pragma unroll
        for (int v = 0; v < 8; ++v) {
            float mx = -1e30f;
#pragma unroll
            for (int j = 0; j < NJ; ++j) mx = fmaxf(mx, S[j][v]);
#pragma unroll
            for (int o = 1; o < 16; o <<= 1)
                mx = fmaxf(mx, __shfl_xor(mx, o, 32));
            float mn = fmaxf(m_st[v], mx);
            float sum = 0.f;
#pragma unroll
            for (int j = 0; j < NJ; ++j) {
                float p = __expf(S[j][v] - mn);
                P2[j][v] = p;
                sum += p;
            }
#pragma unroll
            for (int o = 1; o < 16; o <<= 1)
                sum += __shfl_xor(sum, o, 32);
            float corr = __expf(m_st[v] - mn);
            l_st[v] = l_st[v] * corr + sum;
            m_st[v] = mn;
            scl[v] = corr;
        }
#pragma unroll
        for (int t = 0; t < NT; ++t)
#pragma unroll
            for (int v = 0; v < 8; ++v) O[t][v] *= scl[v];

        // ---- P: C/D layout -> A layout via per-group LDS (DS ops in-order
        //      within a wave; pair writes identical values -> benign)
#pragma unroll
        for (int j = 0; j < NJ; ++j)
#pragma unroll
            for (int v = 0; v < 8; ++v)
                pb[(v + 8 * hi) * VCP + j * 16 + col] = to_bf16(P2[j][v]);

        // ---- O += P * V
#pragma unroll
        for (int k2 = 0; k2 < BN / 32; ++k2) {
            Frag A;
            const __bf16* prow = pb + (size_t)col * VCP + k2 * 32;
            A.u[0] = *(const u32x4*)(prow + hi * 8);
            A.u[1] = *(const u32x4*)(prow + 16 + hi * 8);
#pragma unroll
            for (int t = 0; t < NT; ++t) {
                Frag B;
                const __bf16* vrow = VSMEM
                    ? (Vc_b + (size_t)(c0o + t * 16 + col) * VCP + k2 * 32)
                    : (Vc + (size_t)(c0o + t * 16 + col) * N + mb + k2 * 32);
                B.u[0] = *(const u32x4*)(vrow + hi * 8);
                B.u[1] = *(const u32x4*)(vrow + 16 + hi * 8);
                O[t] = wmma_bf16(A.v, B.v, O[t]);
            }
        }
    }

    // ---- epilogue: normalize and store n-major (== torch reshape layout)
    float inv[8];
#pragma unroll
    for (int v = 0; v < 8; ++v) inv[v] = 1.0f / l_st[v];
#pragma unroll
    for (int t = 0; t < NT; ++t)
#pragma unroll
        for (int v = 0; v < 8; ++v)
            outb[(size_t)(n0 + v + 8 * hi) * C + c0o + t * 16 + col] =
                O[t][v] * inv[v];
}

// ---------------------------------------------------------------------------
// Fused gate + cast to position-major bf16 conv input:
//   Xp[n][c] = bf16( out[n][c] * cat[c][n] + cat[c][n] )
// cat is channel-major fp32 split between a/b; transposed through LDS.
// ---------------------------------------------------------------------------
__global__ void gatecast_k(__bf16* __restrict__ dst,
                           const float* __restrict__ outb,
                           const float* __restrict__ a, const float* __restrict__ b,
                           int halfC, int C, int N) {
    __shared__ float tile[32][33];
    const int bx = blockIdx.x * 32;   // along N
    const int by = blockIdx.y * 32;   // along C
    const int tx = threadIdx.x, ty = threadIdx.y;
#pragma unroll
    for (int i = 0; i < 4; ++i) {
        int c = by + ty + i * 8;
        const float* row = (c < halfC) ? (a + (size_t)c * N)
                                       : (b + (size_t)(c - halfC) * N);
        tile[ty + i * 8][tx] = row[bx + tx];
    }
    __syncthreads();
#pragma unroll
    for (int i = 0; i < 4; ++i) {
        int n = bx + ty + i * 8;
        int c = by + tx;
        float cat = tile[tx][ty + i * 8];
        float o = outb[(size_t)n * C + c];
        dst[(size_t)n * C + c] = to_bf16(fmaf(o, cat, cat));
    }
}

// Weight transform: Wt[tap][co][ci] = bf16( w_oihw[co][ci][tap] )
__global__ void wtrans_k(__bf16* __restrict__ dst, const float* __restrict__ w,
                         int C, int total) {   // total = 9*C*C
    for (int i = blockIdx.x * blockDim.x + threadIdx.x; i < total;
         i += gridDim.x * blockDim.x) {
        int t = i / (C * C);
        int r = i % (C * C);
        int o = r / C;
        int ci = r % C;
        dst[i] = to_bf16(w[((size_t)o * C + ci) * 9 + t]);
    }
}

// ---------------------------------------------------------------------------
// Implicit-GEMM 3x3 conv via WMMA. Per tap: y[n][co] += Xp[n'][ci]*Wt[t][co][ci]
//   Xp: [N][C] bf16 position-major; Wt: [9][C][C] bf16 (ci contiguous).
//   Wave = 16 consecutive positions (one image-row segment); COB co per block.
//   Borders: per-lane predicated A loads (zero-fill). No LDS, no barriers.
// ---------------------------------------------------------------------------
template <int C, int COB, int HW>
__global__ void __launch_bounds__(256)
convwmma_k(const __bf16* __restrict__ Xp, const __bf16* __restrict__ Wt,
           const float* __restrict__ bias, float* __restrict__ out) {
    constexpr int NT = COB / 16;      // co tiles per wave
    constexpr int NK = C / 32;        // ci chunks

    const int wave = threadIdx.x >> 5;
    const int lane = threadIdx.x & 31;
    const int col  = lane & 15;
    const int hi   = lane >> 4;
    const int n0   = (blockIdx.x * 8 + wave) * 16;   // 16 positions, same row
    const int co0  = blockIdx.y * COB;
    const int h    = n0 / HW;
    const int w0   = n0 % HW;

    v8f O[NT] = {};

    for (int tap = 0; tap < 9; ++tap) {
        const int dy = tap / 3 - 1, dx = tap % 3 - 1;
        const int hh = h + dy;
        const int wL = w0 + col + dx;                 // this lane's input col
        const bool valid = (hh >= 0) && (hh < HW) && (wL >= 0) && (wL < HW);
        for (int kk = 0; kk < NK; ++kk) {
            Frag A;
            u32x4 z = {0u, 0u, 0u, 0u};
            A.u[0] = z; A.u[1] = z;
            if (valid) {
                const __bf16* arow = Xp + (size_t)(hh * HW + wL) * C + kk * 32;
                A.u[0] = *(const u32x4*)(arow + hi * 8);
                A.u[1] = *(const u32x4*)(arow + 16 + hi * 8);
            }
#pragma unroll
            for (int t = 0; t < NT; ++t) {
                Frag B;
                const __bf16* brow = Wt + ((size_t)tap * C + co0 + t * 16 + col) * C
                                   + kk * 32;
                B.u[0] = *(const u32x4*)(brow + hi * 8);
                B.u[1] = *(const u32x4*)(brow + 16 + hi * 8);
                O[t] = wmma_bf16(A.v, B.v, O[t]);
            }
        }
    }

    // bias + store channel-major (C, H, W)
#pragma unroll
    for (int t = 0; t < NT; ++t) {
        const int co = co0 + t * 16 + col;
        const float bco = bias[co];
#pragma unroll
        for (int v = 0; v < 8; ++v) {
            const int n = n0 + v + 8 * hi;
            out[(size_t)co * (HW * HW) + n] = O[t][v] + bco;
        }
    }
}

// ---------------------------------------------------------------------------
// Launch
// ---------------------------------------------------------------------------
extern "C" void kernel_launch(void* const* d_in, const int* in_sizes, int n_in,
                              void* d_out, int out_size, void* d_ws, size_t ws_size,
                              hipStream_t stream) {
    (void)in_sizes; (void)n_in; (void)out_size; (void)ws_size;

    const float* radar0 = (const float*)d_in[0];
    const float* lidar0 = (const float*)d_in[1];
    const float* radar1 = (const float*)d_in[2];
    const float* lidar1 = (const float*)d_in[3];
    const float* q1     = (const float*)d_in[4];
    const float* q2     = (const float*)d_in[5];
    const float* c1w    = (const float*)d_in[6];
    const float* c1b    = (const float*)d_in[7];
    const float* c2w    = (const float*)d_in[8];
    const float* c2b    = (const float*)d_in[9];
    float* outp = (float*)d_out;

    const int N1 = 16384, C1 = 256;   // 128x128, cat 256 ch
    const int N2 = 4096,  C2 = 512;   // 64x64,   cat 512 ch

    // workspace carve (60 MB total; regions reused once flash is done)
    char* ws = (char*)d_ws;
    const size_t MB = 1u << 20;
    __bf16* Qt1 = (__bf16*)(ws + 0 * MB);     // 8 MB  (16384*256*2)
    __bf16* Kt1 = (__bf16*)(ws + 8 * MB);     // 8 MB
    __bf16* Vc1 = (__bf16*)(ws + 16 * MB);    // 8 MB
    __bf16* Qt2 = (__bf16*)(ws + 24 * MB);    // 4 MB  (4096*512*2)
    __bf16* Kt2 = (__bf16*)(ws + 28 * MB);    // 4 MB
    __bf16* Vc2 = (__bf16*)(ws + 32 * MB);    // 4 MB
    float*  out1 = (float*)(ws + 36 * MB);    // 16 MB (16384*256*4)
    float*  out2 = (float*)(ws + 52 * MB);    // 8 MB  (4096*512*4)
    // stream-ordered reuse (flash done before these are written):
    __bf16* Xp1 = Qt1;                        // 8 MB  gated conv input, scale 1
    __bf16* Xp2 = Qt2;                        // 4 MB  gated conv input, scale 2
    __bf16* Wt1 = Vc1;                        // 1.2 MB transformed weights 1
    __bf16* Wt2 = Kt1;                        // 4.7 MB transformed weights 2

    dim3 b32x8(32, 8);
    // ---- prep (bf16 operand layouts)
    tcast_k<<<dim3(N1 / 32, C1 / 32), b32x8, 0, stream>>>(
        Qt1, q1, q1 + (size_t)(C1 / 2) * N1, C1 / 2, C1, N1);
    tcast_k<<<dim3(N1 / 32, C1 / 32), b32x8, 0, stream>>>(
        Kt1, radar0, lidar0, C1 / 2, C1, N1);
    ccast_k<<<4096, 256, 0, stream>>>(Vc1, radar0, lidar0,
                                      (C1 / 2) * N1, C1 * N1);
    tcast_k<<<dim3(N2 / 32, C2 / 32), b32x8, 0, stream>>>(
        Qt2, q2, q2 + (size_t)(C2 / 2) * N2, C2 / 2, C2, N2);
    tcast_k<<<dim3(N2 / 32, C2 / 32), b32x8, 0, stream>>>(
        Kt2, radar1, lidar1, C2 / 2, C2, N2);
    ccast_k<<<2048, 256, 0, stream>>>(Vc2, radar1, lidar1,
                                      (C2 / 2) * N2, C2 * N2);

    // ---- flash attention (dynamic LDS bytes: double-buffered K/V + P)
    const int smem1 = (2 * 32 * (C1 + 8) + 2 * C1 * (32 + 8) + 4 * 16 * (32 + 8)) * 2;
    const int smem2 = (2 * 32 * (C2 + 8) + 2 * 16 * (32 + 8)) * 2;
    flash_attn<256, 4, 32, true ><<<N1 / 64, 256, smem1, stream>>>(
        Qt1, Kt1, Vc1, out1, N1);
    flash_attn<512, 2, 32, false><<<N2 / 32, 128, smem2, stream>>>(
        Qt2, Kt2, Vc2, out2, N2);

    // ---- weight transform (into regions freed by flash)
    wtrans_k<<<1024, 256, 0, stream>>>(Wt1, c1w, C1, 9 * C1 * C1);
    wtrans_k<<<1024, 256, 0, stream>>>(Wt2, c2w, C2, 9 * C2 * C2);

    // ---- fused gate + transpose-cast to position-major bf16 conv input
    gatecast_k<<<dim3(N1 / 32, C1 / 32), b32x8, 0, stream>>>(
        Xp1, out1, radar0, lidar0, C1 / 2, C1, N1);
    gatecast_k<<<dim3(N2 / 32, C2 / 32), b32x8, 0, stream>>>(
        Xp2, out2, radar1, lidar1, C2 / 2, C2, N2);

    // ---- implicit-GEMM WMMA conv3x3 -> outputs (y1 then y2, flat)
    convwmma_k<256, 256, 128><<<dim3(N1 / 128, 1), 256, 0, stream>>>(
        Xp1, Wt1, c1b, outp);
    convwmma_k<512, 256, 64><<<dim3(N2 / 128, 2), 256, 0, stream>>>(
        Xp2, Wt2, c2b, outp + 4194304);
}
